// GConvLSTM_W__42691974922289
// MI455X (gfx1250) — matrix operand order repaired and checked by reading, not compiled, not run on hardware
//
#include <hip/hip_runtime.h>
#include <hip/hip_bf16.h>

// ---------------------------------------------------------------------------
// GConvLSTM (ChebConv K=3, lambda_max=2) for MI455X / gfx1250.
// lambda_max=2 => Lhat(v) = -A v, so T0=x, T1=-Ax, T2=2A^2x - x and every
// ChebConv is x@(W0-W2) + (Ax)@(-W1) + (A^2x)@(2W2) + b.  All 8 convs fuse
// into one [N,384]x[384,256] GEMM done with v_wmma_f32_16x16x32_bf16, with
// weight tiles staged into LDS via GLOBAL_LOAD_ASYNC_TO_LDS when available.
// ---------------------------------------------------------------------------

typedef __attribute__((ext_vector_type(16))) __bf16 v16bf;
typedef __attribute__((ext_vector_type(8)))  __bf16 v8bf;
typedef __attribute__((ext_vector_type(8)))  float  v8f;
typedef __attribute__((ext_vector_type(4)))  float  v4f;
typedef __attribute__((vector_size(16)))     int    v4i32;

#if defined(__has_builtin)
#if __has_builtin(__builtin_amdgcn_global_load_async_to_lds_b128)
#define USE_ASYNC_LDS 1
#endif
#if __has_builtin(__builtin_amdgcn_s_wait_asynccnt)
#define HAVE_WAIT_ASYNC_BUILTIN 1
#endif
#endif

// ---------------- simple utility kernels ----------------

__global__ void zero_kernel(float* __restrict__ p, long long n) {
  long long i = (long long)blockIdx.x * blockDim.x + threadIdx.x;
  if (i < n) p[i] = 0.f;
}

__global__ void deg_kernel(const int* __restrict__ row, const float* __restrict__ w,
                           float* __restrict__ deg, int E) {
  int e = blockIdx.x * blockDim.x + threadIdx.x;
  if (e < E) atomicAdd(deg + row[e], w[e]);
}

__global__ void dinv_kernel(float* __restrict__ deg, int n) {
  int i = blockIdx.x * blockDim.x + threadIdx.x;
  if (i < n) {
    float d = deg[i];
    deg[i] = (d > 0.f) ? rsqrtf(d) : 0.f;
  }
}

__global__ void nw_kernel(const int* __restrict__ row, const int* __restrict__ col,
                          const float* __restrict__ w, const float* __restrict__ dinv,
                          float* __restrict__ nw, int E) {
  int e = blockIdx.x * blockDim.x + threadIdx.x;
  if (e < E) nw[e] = dinv[row[e]] * w[e] * dinv[col[e]];
}

// out[row] += nw * in[col]   (64 features, 2 per thread, 32 threads per edge)
__global__ void spmm_kernel(const int* __restrict__ row, const int* __restrict__ col,
                            const float* __restrict__ nw,
                            const float* __restrict__ vin, float* __restrict__ vout, int E) {
  int gid = blockIdx.x * blockDim.x + threadIdx.x;
  int e = gid >> 5;
  if (e >= E) return;
  int f = (gid & 31) * 2;
  int r = row[e], c = col[e];
  float w = nw[e];
  const float* src = vin + (size_t)c * 64 + f;
  float v0 = src[0], v1 = src[1];
  atomicAdd(vout + (size_t)r * 64 + f + 0, w * v0);
  atomicAdd(vout + (size_t)r * 64 + f + 1, w * v1);
}

// Build combined weight matrix WbigT[256 cols][384 rows] (bf16, K-major per
// column so B-fragments are contiguous) and combined bias[256].
// Row blocks (K dim): 0:X 1:AX 2:AAX 3:H 4:AH 5:AAH.  Col blocks: gates i,f,c,o.
__global__ void prep_w_kernel(
    const float* Wxi, const float* bxi, const float* Whi, const float* bhi,
    const float* Wxf, const float* bxf, const float* Whf, const float* bhf,
    const float* Wxc, const float* bxc, const float* Whc, const float* bhc,
    const float* Wxo, const float* bxo, const float* Who, const float* bho,
    const float* bi,  const float* bf,  const float* bc,  const float* bo,
    __bf16* __restrict__ WbigT, float* __restrict__ biasc) {
  int idx = blockIdx.x * blockDim.x + threadIdx.x;
  if (idx >= 256 * 384) return;
  int c = idx / 384, r = idx % 384;
  int g = c >> 6, cout = c & 63;
  int sb = r / 64, cin = r % 64;
  const float* Wx[4] = {Wxi, Wxf, Wxc, Wxo};
  const float* Wh[4] = {Whi, Whf, Whc, Who};
  const float* W = (sb < 3) ? Wx[g] : Wh[g];
  int term = (sb < 3) ? sb : sb - 3;
  int o = cin * 64 + cout;
  float v;
  if (term == 0)      v = W[o] - W[2 * 4096 + o];   // x@(W0 - W2)
  else if (term == 1) v = -W[4096 + o];             // (Ax)@(-W1)
  else                v = 2.f * W[2 * 4096 + o];    // (A^2 x)@(2 W2)
  WbigT[(size_t)c * 384 + r] = (__bf16)v;
  if (r == 0) {
    const float* bx[4] = {bxi, bxf, bxc, bxo};
    const float* bh[4] = {bhi, bhf, bhc, bho};
    const float* bg[4] = {bi,  bf,  bc,  bo};
    biasc[c] = bx[g][cout] + bh[g][cout] + bg[g][cout];
  }
}

// ---------------- fused WMMA LSTM-cell kernel ----------------
// 1 wave = 16 nodes, 8 waves per block. A = 16x384 (6 fp32 sources cvt->bf16),
// B = WbigT tiles staged in LDS (async copy), 192 WMMAs per tile, gate math.

__device__ __forceinline__ float sigm_f(float x) { return 1.f / (1.f + __expf(-x)); }
__device__ __forceinline__ float tanh_f(float x) { return 2.f / (1.f + __expf(-2.f * x)) - 1.f; }

__global__ __launch_bounds__(256) void cell_kernel(
    const float* __restrict__ X,  const float* __restrict__ H,  const float* __restrict__ C,
    const float* __restrict__ AX, const float* __restrict__ AAX,
    const float* __restrict__ AH, const float* __restrict__ AAH,
    const __bf16* __restrict__ WbigT, const float* __restrict__ biasc,
    float* __restrict__ Hout, float* __restrict__ Cout, int ntiles) {
  __shared__ __bf16 shW[64 * 384];  // 48 KB: weights for one 16-feature slab, all 4 gates

  const int wave = threadIdx.x >> 5;
  const int lane = threadIdx.x & 31;
  const int tile = blockIdx.x * 8 + wave;
  const bool active = (tile < ntiles);
  const int m = lane & 15;       // A row / D column lane
  const int hi = lane >> 4;      // lane half selects K sub-range / D row offset
  const int n = m;
  const int row0 = tile * 16;

  // ---- A fragments: 12 chunks of K=32 over [X|AX|AAX|H|AH|AAH] ----
  // 16-bit A 16x32 layout: lane<16  -> K = {0..7, 16..23}
  //                        lane>=16 -> K = {8..15, 24..31}
  const float* srcs[6] = {X, AX, AAX, H, AH, AAH};
  v16bf a[12];
  if (active) {
#pragma unroll
    for (int kk = 0; kk < 12; ++kk) {
      const float* sp = srcs[kk >> 1] + (size_t)(row0 + m) * 64 + (kk & 1) * 32 + hi * 8;
      v8f f01 = *(const v8f*)(sp);       // K elements 0..7
      v8f f23 = *(const v8f*)(sp + 16);  // K elements 16..23
      v8bf lo  = __builtin_convertvector(f01, v8bf);
      v8bf hi8 = __builtin_convertvector(f23, v8bf);
      a[kk] = __builtin_shufflevector(lo, hi8, 0, 1, 2, 3, 4, 5, 6, 7,
                                      8, 9, 10, 11, 12, 13, 14, 15);
    }
  }

  // ---- 4 feature slabs of 16 output features; all 4 gates per slab ----
  for (int t = 0; t < 4; ++t) {
    __syncthreads();
    // stage weight columns {g*64 + t*16 + nn : g in 0..3, nn in 0..15} into LDS
    for (int ch = threadIdx.x; ch < 3072; ch += 256) {
      int lc = ch / 48;        // local column 0..63 (= g*16 + nn)
      int part = ch % 48;      // 8 bf16 (16 B) per chunk along K
      int cw = (lc >> 4) * 64 + t * 16 + (lc & 15);
#if defined(USE_ASYNC_LDS)
      __builtin_amdgcn_global_load_async_to_lds_b128(
          (__attribute__((address_space(1))) v4i32*)(WbigT + (size_t)cw * 384 + part * 8),
          (__attribute__((address_space(3))) v4i32*)(shW + lc * 384 + part * 8),
          0, 0);
#else
      *(v8bf*)(shW + lc * 384 + part * 8) =
          *(const v8bf*)(WbigT + (size_t)cw * 384 + part * 8);
#endif
    }
#if defined(USE_ASYNC_LDS)
#if defined(HAVE_WAIT_ASYNC_BUILTIN)
    __builtin_amdgcn_s_wait_asynccnt(0);
#else
    asm volatile("s_wait_asynccnt 0x0" ::: "memory");
#endif
#endif
    __syncthreads();

    if (active) {
      v8f acc[4] = {};
#pragma unroll
      for (int g = 0; g < 4; ++g) {
#pragma unroll
        for (int kk = 0; kk < 12; ++kk) {
          // 16-bit B 32x16 layout: lane<16 -> K {0..7, 16..23}, lane>=16 -> K {8..15, 24..31}
          const __bf16* bp = shW + (g * 16 + n) * 384 + kk * 32 + hi * 8;
          v8bf blo = *(const v8bf*)(bp);
          v8bf bhi = *(const v8bf*)(bp + 16);
          v16bf b = __builtin_shufflevector(blo, bhi, 0, 1, 2, 3, 4, 5, 6, 7,
                                            8, 9, 10, 11, 12, 13, 14, 15);
          acc[g] = __builtin_amdgcn_wmma_f32_16x16x32_bf16(
              false, a[kk], false, b, (short)0, acc[g], false, false);
        }
      }

      // ---- gate math for features t*16 + n, rows row0 + r + 8*hi ----
      float bi_ = biasc[0 * 64 + t * 16 + n];
      float bf_ = biasc[1 * 64 + t * 16 + n];
      float bc_ = biasc[2 * 64 + t * 16 + n];
      float bo_ = biasc[3 * 64 + t * 16 + n];
#pragma unroll
      for (int r = 0; r < 8; ++r) {
        size_t idx = (size_t)(row0 + r + 8 * hi) * 64 + t * 16 + n;
        float I = sigm_f(acc[0][r] + bi_);
        float F = sigm_f(acc[1][r] + bf_);
        float T = tanh_f(acc[2][r] + bc_);
        float O = sigm_f(acc[3][r] + bo_);
        float Cn = F * C[idx] + I * T;
        Hout[idx] = O * tanh_f(Cn);
        Cout[idx] = Cn;
      }
    }
  }
}

// ---------------------------------------------------------------------------

extern "C" void kernel_launch(void* const* d_in, const int* in_sizes, int n_in,
                              void* d_out, int out_size, void* d_ws, size_t ws_size,
                              hipStream_t stream) {
  const float* X  = (const float*)d_in[0];
  const int* eidx = (const int*)d_in[1];
  const float* ew = (const float*)d_in[2];
  const float* H  = (const float*)d_in[3];
  const float* C  = (const float*)d_in[4];

  const int N = in_sizes[0] / 64;
  const int E = in_sizes[2];
  const int* rows = eidx;
  const int* cols = eidx + E;

  // workspace layout (floats)
  float* ws   = (float*)d_ws;
  float* deg  = ws;                                 // N (becomes dinv)
  float* nw   = deg + N;                            // E
  float* AX   = nw + E;                             // N*64
  float* AAX  = AX  + (size_t)N * 64;
  float* AH   = AAX + (size_t)N * 64;
  float* AAH  = AH  + (size_t)N * 64;
  float* biasc = AAH + (size_t)N * 64;              // 256
  __bf16* WbigT = (__bf16*)(biasc + 256);           // 256*384 bf16

  float* Hout = (float*)d_out;
  float* Cout = Hout + (size_t)N * 64;

  long long nz = (long long)N + E + 4LL * N * 64;
  zero_kernel<<<(int)((nz + 255) / 256), 256, 0, stream>>>(ws, nz);

  prep_w_kernel<<<(256 * 384 + 255) / 256, 256, 0, stream>>>(
      (const float*)d_in[5],  (const float*)d_in[6],  (const float*)d_in[7],  (const float*)d_in[8],
      (const float*)d_in[9],  (const float*)d_in[10], (const float*)d_in[11], (const float*)d_in[12],
      (const float*)d_in[13], (const float*)d_in[14], (const float*)d_in[15], (const float*)d_in[16],
      (const float*)d_in[17], (const float*)d_in[18], (const float*)d_in[19], (const float*)d_in[20],
      (const float*)d_in[21], (const float*)d_in[22], (const float*)d_in[23], (const float*)d_in[24],
      WbigT, biasc);

  deg_kernel<<<(E + 255) / 256, 256, 0, stream>>>(rows, ew, deg, E);
  dinv_kernel<<<(N + 255) / 256, 256, 0, stream>>>(deg, N);
  nw_kernel<<<(E + 255) / 256, 256, 0, stream>>>(rows, cols, ew, deg, nw, E);

  int spmm_blocks = (int)(((long long)E * 32 + 255) / 256);
  spmm_kernel<<<spmm_blocks, 256, 0, stream>>>(rows, cols, nw, X,  AX,  E);
  spmm_kernel<<<spmm_blocks, 256, 0, stream>>>(rows, cols, nw, AX, AAX, E);
  spmm_kernel<<<spmm_blocks, 256, 0, stream>>>(rows, cols, nw, H,  AH,  E);
  spmm_kernel<<<spmm_blocks, 256, 0, stream>>>(rows, cols, nw, AH, AAH, E);

  int ntiles = N / 16;  // N = 50000 -> 3125 tiles, 1 wave each
  cell_kernel<<<(ntiles + 7) / 8, 256, 0, stream>>>(
      X, H, C, AX, AAX, AH, AAH, WbigT, biasc, Hout, Cout, ntiles);
}
